// SimpleRotRNN_57277683859719
// MI455X (gfx1250) — matrix operand order repaired and checked by reading, not compiled, not run on hardware
//
#include <hip/hip_runtime.h>
#include <hip/hip_bf16.h>
#include <math.h>

typedef __attribute__((ext_vector_type(16))) __bf16 v16bf;
typedef __attribute__((ext_vector_type(8)))  float  v8f;

#define HH 64
#define NNE 16
#define MD 1024
#define TT 16384
#define CHUNK 128
#define NCHUNK (TT / CHUNK)

// ---------------- per-head stats: gamma, xi ----------------
__global__ void pre_stats(const float* __restrict__ B, const float* __restrict__ gamma_log,
                          float* __restrict__ gamma_ws, float* __restrict__ xi_ws) {
  int h = blockIdx.x;
  __shared__ float red[256];
  const float* bh = B + (size_t)h * NNE * MD;
  float s = 0.f;
  for (int i = threadIdx.x; i < NNE * MD; i += 256) { float v = bh[i]; s += v * v; }
  red[threadIdx.x] = s; __syncthreads();
  for (int o = 128; o > 0; o >>= 1) {
    if (threadIdx.x < o) red[threadIdx.x] += red[threadIdx.x + o];
    __syncthreads();
  }
  if (threadIdx.x == 0) {
    float g = expf(-expf(gamma_log[h]));
    gamma_ws[h] = g;
    xi_ws[h] = sqrtf((1.f - g * g) / red[0]);
  }
}

// ---------------- per-head expm(M - M^T), 16x16, scale&square s=8 ----------------
__global__ void pre_expm(const float* __restrict__ M, float* __restrict__ P) {
  int h = blockIdx.x; int tid = threadIdx.x;
  int i = tid >> 4, j = tid & 15;
  __shared__ float As[256], Tc[256], Es[256];
  float a = (M[(size_t)h * 256 + i * 16 + j] - M[(size_t)h * 256 + j * 16 + i]) * (1.0f / 256.0f);
  As[tid] = a;
  float iij = (i == j) ? 1.f : 0.f;
  Tc[tid] = iij;
  float e = iij;
  __syncthreads();
  for (int n = 1; n <= 12; n++) {          // Taylor on A/256
    float acc = 0.f;
    #pragma unroll
    for (int k = 0; k < 16; k++) acc += Tc[i * 16 + k] * As[k * 16 + j];
    acc *= (1.0f / (float)n);
    __syncthreads();
    Tc[tid] = acc; e += acc;
    __syncthreads();
  }
  Es[tid] = e; __syncthreads();
  for (int s = 0; s < 8; s++) {            // square 8 times
    float acc = 0.f;
    #pragma unroll
    for (int k = 0; k < 16; k++) acc += Es[i * 16 + k] * Es[k * 16 + j];
    __syncthreads();
    Es[tid] = acc;
    __syncthreads();
  }
  P[(size_t)h * 256 + tid] = Es[tid];
}

// ---------------- fold P into weights, store bf16 hi/lo split directly ----------------
__global__ void pre_weights(const float* __restrict__ B, const float* __restrict__ C,
                            const float* __restrict__ P, const float* __restrict__ xi,
                            __bf16* __restrict__ W1h, __bf16* __restrict__ W1l,
                            __bf16* __restrict__ W2h, __bf16* __restrict__ W2l) {
  int idx = blockIdx.x * blockDim.x + threadIdx.x;   // 2M threads
  int sel = idx >> 20; int r = idx & 0xFFFFF;
  float acc = 0.f;
  if (sel == 0) {
    int mo = r >> 10, d = r & 1023;
    int h = mo >> 4, n = mo & 15;
    const float* Ph = P + h * 256;
    const float* Bh = B + (size_t)h * NNE * MD + d;
    #pragma unroll
    for (int m = 0; m < 16; m++) acc += Ph[n * 16 + m] * Bh[(size_t)m * MD];
    acc *= xi[h];
    __bf16 hv = (__bf16)acc;
    W1h[r] = hv; W1l[r] = (__bf16)(acc - (float)hv);
  } else {
    int d = r >> 10, col = r & 1023;
    int h = col >> 4, n = col & 15;
    const float* Ph = P + h * 256;
    const float* Cr = C + (size_t)d * MD + h * 16;
    #pragma unroll
    for (int m = 0; m < 16; m++) acc += Cr[m] * Ph[m * 16 + n];
    __bf16 hv = (__bf16)acc;
    W2h[r] = hv; W2l[r] = (__bf16)(acc - (float)hv);
  }
}

// ---------------- split f32 array into bf16 hi/lo arrays ----------------
__global__ void split_f32(const float* __restrict__ src, __bf16* __restrict__ hi,
                          __bf16* __restrict__ lo) {
  int i = (blockIdx.x * 256 + threadIdx.x) * 8;
  float4 a = *(const float4*)(src + i);
  float4 b = *(const float4*)(src + i + 4);
  float v[8] = {a.x, a.y, a.z, a.w, b.x, b.y, b.z, b.w};
  #pragma unroll
  for (int j = 0; j < 8; j++) {
    __bf16 h = (__bf16)v[j];
    hi[i + j] = h;
    lo[i + j] = (__bf16)(v[j] - (float)h);
  }
}

// ---------------- fragment loaders matching WMMA VGPR layouts ----------------
__device__ __forceinline__ v16bf load_afrag(const __bf16* rowp, int kb, int hf) {
  union { v16bf v; uint4 q[2]; } f;
  f.q[0] = *(const uint4*)(rowp + kb + 8 * hf);        // K = 8*hf + 0..7
  f.q[1] = *(const uint4*)(rowp + kb + 16 + 8 * hf);   // K = 16 + 8*hf + 0..7
  return f.v;
}
__device__ __forceinline__ v16bf load_bfrag(const __bf16* rowp, int kb, int hf) {
  union { v16bf v; uint4 q[2]; } f;
  const __bf16* p = rowp + kb + 8 * hf;                // K = 8*hf + 0..7, 16 + 8*hf + 0..7
  f.q[0] = *(const uint4*)(p);
  f.q[1] = *(const uint4*)(p + 16);
  return f.v;
}

#define WMMA3(ACC, AH, AL, BH, BL)                                                            \
  ACC = __builtin_amdgcn_wmma_f32_16x16x32_bf16(false, AL, false, BH, (short)0, ACC, false, false); \
  ACC = __builtin_amdgcn_wmma_f32_16x16x32_bf16(false, AH, false, BL, (short)0, ACC, false, false); \
  ACC = __builtin_amdgcn_wmma_f32_16x16x32_bf16(false, AH, false, BH, (short)0, ACC, false, false);

// ---------------- WMMA GEMM: Out[t,m] = sum_k W[m,k] In[t,k]  (+ D[m]*U[t,m]) ----------------
// Pure-bf16 inner loop (pre-split operands), 2x4 register tile (32t x 64m) per wave:
// 24 WMMAs per 24 b128 loads per K=32 step.
__global__ __launch_bounds__(256) void gemm_wmma_split(
    const __bf16* __restrict__ Wh, const __bf16* __restrict__ Wl,
    const __bf16* __restrict__ Ah, const __bf16* __restrict__ Al,
    float* __restrict__ Out, const float* __restrict__ Dvec,
    const float* __restrict__ U, int fuseD) {
  int lane = threadIdx.x & 31;
  int wave = threadIdx.x >> 5;
  int hf   = lane >> 4;
  int l16  = lane & 15;
  int mbase = blockIdx.x * 64;
  int tbase = blockIdx.y * 256 + wave * 32;

  const __bf16* arh = Ah + (size_t)(tbase + l16) * MD;
  const __bf16* arl = Al + (size_t)(tbase + l16) * MD;
  const __bf16* brh = Wh + (size_t)(mbase + l16) * MD;
  const __bf16* brl = Wl + (size_t)(mbase + l16) * MD;

  v8f acc[2][4];
  #pragma unroll
  for (int i = 0; i < 2; i++)
    #pragma unroll
    for (int j = 0; j < 4; j++) acc[i][j] = (v8f){0.f,0.f,0.f,0.f,0.f,0.f,0.f,0.f};

  for (int kb = 0; kb < MD; kb += 32) {
    v16bf fah[2], fal[2], fbh[4], fbl[4];
    #pragma unroll
    for (int i = 0; i < 2; i++) {
      fah[i] = load_afrag(arh + (size_t)(16 * i) * MD, kb, hf);
      fal[i] = load_afrag(arl + (size_t)(16 * i) * MD, kb, hf);
    }
    #pragma unroll
    for (int j = 0; j < 4; j++) {
      fbh[j] = load_bfrag(brh + (size_t)(16 * j) * MD, kb, hf);
      fbl[j] = load_bfrag(brl + (size_t)(16 * j) * MD, kb, hf);
    }
    if (kb + 32 < MD) {
      __builtin_prefetch(arh + kb + 32, 0, 1);   // global_prefetch_b8
      __builtin_prefetch(brh + kb + 32, 0, 1);
    }
    #pragma unroll
    for (int i = 0; i < 2; i++)
      #pragma unroll
      for (int j = 0; j < 4; j++) {
        WMMA3(acc[i][j], fah[i], fal[i], fbh[j], fbl[j]);
      }
  }

  #pragma unroll
  for (int ts = 0; ts < 2; ts++) {
    #pragma unroll
    for (int ms = 0; ms < 4; ms++) {
      int m = mbase + 16 * ms + l16;
      float dm = fuseD ? Dvec[m] : 0.f;
      #pragma unroll
      for (int r = 0; r < 8; r++) {
        int t = tbase + 16 * ts + r + 8 * hf;     // C/D layout: VGPR r -> M = r + 8*half
        float v = acc[ts][ms][r];
        if (fuseD) v += dm * U[(size_t)t * MD + m];
        Out[(size_t)t * MD + m] = v;
      }
    }
  }
}

// ---------------- scan pass 1: per-chunk local scan (inclusive), record chunk ends ----------------
__global__ void scan_local(float* __restrict__ X, const float* __restrict__ theta,
                           const float* __restrict__ gamma_ws, float* __restrict__ ends) {
  int p = threadIdx.x;               // 512 pairs
  int chunk = blockIdx.x;
  float th = theta[p];
  float g  = gamma_ws[p >> 3];
  float gc = g * cosf(th), gs = g * sinf(th);
  float sa = 0.f, sb = 0.f;
  size_t base = (size_t)chunk * CHUNK * MD + 2 * p;
  for (int i = 0; i < CHUNK; i++) {
    float2 x = *(const float2*)(X + base + (size_t)i * MD);
    float na = gc * sa - gs * sb + x.x;
    float nb = gs * sa + gc * sb + x.y;
    sa = na; sb = nb;
    *(float2*)(X + base + (size_t)i * MD) = make_float2(sa, sb);
  }
  ends[chunk * MD + 2 * p]     = sa;
  ends[chunk * MD + 2 * p + 1] = sb;
}

// ---------------- scan pass 2: exclusive carry scan across chunks (A^L = g^L R(L*theta)) ----------
__global__ void scan_carry(const float* __restrict__ ends, const float* __restrict__ theta,
                           const float* __restrict__ gamma_ws, float* __restrict__ carry) {
  int p = blockIdx.x * blockDim.x + threadIdx.x;
  if (p >= 512) return;
  float th = theta[p];
  float g  = gamma_ws[p >> 3];
  float gl = powf(g, (float)CHUNK);
  float ang = th * (float)CHUNK;
  float gc = gl * cosf(ang), gs = gl * sinf(ang);
  float ca = 0.f, cb = 0.f;
  for (int k = 0; k < NCHUNK; k++) {
    carry[k * MD + 2 * p]     = ca;
    carry[k * MD + 2 * p + 1] = cb;
    float ea = ends[k * MD + 2 * p], eb = ends[k * MD + 2 * p + 1];
    float na = gc * ca - gs * cb + ea;
    float nb = gs * ca + gc * cb + eb;
    ca = na; cb = nb;
  }
}

// ---------------- scan pass 3: s_t = local + A^{dt} carry; emit bf16 hi/lo split --------------
__global__ void scan_apply(const float* __restrict__ X, const float* __restrict__ theta,
                           const float* __restrict__ gamma_ws, const float* __restrict__ carry,
                           __bf16* __restrict__ Xh, __bf16* __restrict__ Xl) {
  int idx = blockIdx.x * 256 + threadIdx.x;   // TT*512 threads
  int p = idx & 511;
  int t = idx >> 9;
  int chunk = t >> 7;
  int dt = (t & (CHUNK - 1)) + 1;
  float ca = carry[chunk * MD + 2 * p], cb = carry[chunk * MD + 2 * p + 1];
  float th = theta[p];
  float g  = gamma_ws[p >> 3];
  float gd = powf(g, (float)dt);
  float ang = th * (float)dt;
  float sn, cs;
  sincosf(ang, &sn, &cs);
  float2 v = *(const float2*)(X + (size_t)t * MD + 2 * p);
  float fa = v.x + gd * (cs * ca - sn * cb);
  float fb = v.y + gd * (sn * ca + cs * cb);
  size_t o = (size_t)t * MD + 2 * p;
  __bf16 ha = (__bf16)fa, hb = (__bf16)fb;
  Xh[o] = ha;     Xh[o + 1] = hb;
  Xl[o] = (__bf16)(fa - (float)ha);
  Xl[o + 1] = (__bf16)(fb - (float)hb);
}

extern "C" void kernel_launch(void* const* d_in, const int* in_sizes, int n_in,
                              void* d_out, int out_size, void* d_ws, size_t ws_size,
                              hipStream_t stream) {
  const float* U         = (const float*)d_in[0];   // (T, 1024)
  const float* theta     = (const float*)d_in[1];   // (64, 8)
  const float* M         = (const float*)d_in[2];   // (64, 16, 16)
  const float* B         = (const float*)d_in[3];   // (64, 16, 1024)
  const float* C         = (const float*)d_in[4];   // (1024, 1024)
  const float* Dv        = (const float*)d_in[5];   // (1024,)
  const float* gamma_log = (const float*)d_in[6];   // (64, 1)
  float* Y = (float*)d_out;

  float* ws       = (float*)d_ws;
  float* gamma_ws = ws;                              // 64
  float* xi       = ws + 64;                         // 64
  float* P        = ws + 128;                        // 64*256
  float* X        = P + 64 * 256;                    // 16M f32
  float* ends     = X + (size_t)TT * MD;             // NCHUNK*1024
  float* carry    = ends + (size_t)NCHUNK * MD;      // NCHUNK*1024
  __bf16* bfb     = (__bf16*)(carry + (size_t)NCHUNK * MD);
  __bf16* W1h = bfb;                                 // 1M bf16 each
  __bf16* W1l = W1h + (size_t)(1 << 20);
  __bf16* W2h = W1l + (size_t)(1 << 20);
  __bf16* W2l = W2h + (size_t)(1 << 20);
  __bf16* Uh  = W2l + (size_t)(1 << 20);             // 16M bf16 each
  __bf16* Ul  = Uh + (size_t)TT * MD;
  __bf16* Xh  = Ul + (size_t)TT * MD;
  __bf16* Xl  = Xh + (size_t)TT * MD;

  pre_stats  <<<HH, 256, 0, stream>>>(B, gamma_log, gamma_ws, xi);
  pre_expm   <<<HH, 256, 0, stream>>>(M, P);
  pre_weights<<<(2 * 1024 * 1024) / 256, 256, 0, stream>>>(B, C, P, xi, W1h, W1l, W2h, W2l);
  split_f32  <<<(TT * MD) / (256 * 8), 256, 0, stream>>>(U, Uh, Ul);

  dim3 gg(MD / 64, TT / 256);
  gemm_wmma_split<<<gg, 256, 0, stream>>>(W1h, W1l, Uh, Ul, X, nullptr, nullptr, 0);

  scan_local <<<NCHUNK, 512, 0, stream>>>(X, theta, gamma_ws, ends);
  scan_carry <<<2, 256, 0, stream>>>(ends, theta, gamma_ws, carry);
  scan_apply <<<(TT * 512) / 256, 256, 0, stream>>>(X, theta, gamma_ws, carry, Xh, Xl);

  gemm_wmma_split<<<gg, 256, 0, stream>>>(W2h, W2l, Xh, Xl, Y, Dv, U, 1);
}
